// self_attention_memory_module_12859132084398
// MI455X (gfx1250) — matrix operand order repaired and checked by reading, not compiled, not run on hardware
//
#include <hip/hip_runtime.h>
#include <cmath>

typedef __attribute__((ext_vector_type(16))) __bf16 v16bf;
typedef __attribute__((ext_vector_type(2)))  __bf16 v2bf;
typedef __attribute__((ext_vector_type(8)))  float  v8f;
typedef __attribute__((ext_vector_type(4)))  unsigned uint4v;
typedef __attribute__((ext_vector_type(8)))  int  int8v;
typedef __attribute__((ext_vector_type(4)))  int  int4v;

typedef unsigned short U16;
typedef unsigned int   U32;

// ---------- helpers ----------

__device__ __forceinline__ U16 f2bfu(float f) {
  U32 u = __builtin_bit_cast(U32, f);
  // round-to-nearest-even bf16
  return (U16)((u + 0x7FFFu + ((u >> 16) & 1u)) >> 16);
}

__device__ __forceinline__ v8f wmma_bf16(v16bf a, v16bf b, v8f c) {
  // D = A(16x32 bf16) x B(32x16 bf16) + C(16x16 f32)
  return __builtin_amdgcn_wmma_f32_16x16x32_bf16(
      /*neg_a=*/false, a, /*neg_b=*/false, b,
      /*c_mod=*/(short)0, c, /*reuse_a=*/false, /*reuse_b=*/false);
}

// A fragment: 16(M) x 32(K), source row-major, row stride ld (elements).
// Lane l: M = l&15, half = l>>4; VGPR v: K = (v>>2)*16 + half*8 + (v&3)*2 (+0,+1)
__device__ __forceinline__ v16bf load_a_frag(const U16* src, int ld, int lane) {
  int mrow = lane & 15;
  int half = lane >> 4;
  const U16* row = src + mrow * ld;
  v16bf a;
#pragma unroll
  for (int v = 0; v < 8; ++v) {
    int k = (v >> 2) * 16 + half * 8 + (v & 3) * 2;
    U32 d = *(const U32*)(row + k);
    v2bf p = __builtin_bit_cast(v2bf, d);
    a[2 * v] = p[0];
    a[2 * v + 1] = p[1];
  }
  return a;
}

// B fragment: 32(K) x 16(N), source stored column-major: element B[k][n] at
// src[n*ld + k] (K contiguous per column). Lane l: N = l&15, half = l>>4;
// VGPR v: K = (v>>2)*16 + half*8 + (v&3)*2 (+0,+1).
__device__ __forceinline__ v16bf load_b_frag(const U16* src, int ld, int lane) {
  int ncol = lane & 15;
  int half = lane >> 4;
  const U16* colp = src + ncol * ld;
  v16bf b;
#pragma unroll
  for (int v = 0; v < 8; ++v) {
    int k = (v >> 2) * 16 + half * 8 + (v & 3) * 2;
    U32 d = *(const U32*)(colp + k);
    v2bf p = __builtin_bit_cast(v2bf, d);
    b[2 * v] = p[0];
    b[2 * v + 1] = p[1];
  }
  return b;
}

__device__ __forceinline__ v8f zero8() {
  v8f z = {0.f, 0.f, 0.f, 0.f, 0.f, 0.f, 0.f, 0.f};
  return z;
}

// ---------- kernel 1: weights -> bf16 ----------
// Wstack rows: [0,64)=Wq [64,128)=Wk [128,192)=Wv [192,256)=Wk2 [256,320)=Wv2
__global__ __launch_bounds__(256) void k_prep_weights(
    const float* Wq, const float* Wk, const float* Wv, const float* Wk2,
    const float* Wv2, const float* Wz, const float* Wm,
    U16* Wstack, U16* Wz16, U16* Wm16) {
  int i = blockIdx.x * blockDim.x + threadIdx.x;
  const int NS = 5 * 4096, NZ = 128 * 128, NM = 192 * 192;
  if (i < NS) {
    int w = i >> 12, r = i & 4095;
    const float* p = (w == 0) ? Wq : (w == 1) ? Wk : (w == 2) ? Wv
                     : (w == 3) ? Wk2 : Wv2;
    Wstack[i] = f2bfu(p[r]);
  } else if (i < NS + NZ) {
    Wz16[i - NS] = f2bfu(Wz[i - NS]);
  } else if (i < NS + NZ + NM) {
    Wm16[i - NS - NZ] = f2bfu(Wm[i - NS - NZ]);
  }
}

// ---------- kernel 2: transpose h,m (B,C,N)->(B,N,C) bf16; h also into Xfin cols 128..191 ----------
__global__ __launch_bounds__(256) void k_transpose_in(
    const float* h, const float* m, U16* ht, U16* mt_, U16* Xfin) {
  const int PER = 16 * 64 * 1024;
  int i = blockIdx.x * blockDim.x + threadIdx.x;
  if (i >= 2 * PER) return;
  bool is_m = i >= PER;
  int j = is_m ? i - PER : i;     // j = ((b*1024)+n)*64 + c
  int b = j >> 16;
  int r = j & 65535;
  int n = r >> 6;
  int c = r & 63;
  float v = (is_m ? m : h)[b * 65536 + c * 1024 + n];
  U16 bv16 = f2bfu(v);
  if (is_m) {
    mt_[j] = bv16;
  } else {
    ht[j] = bv16;
    Xfin[(size_t)((b << 10) + n) * 192 + 128 + c] = bv16;
  }
}

// ---------- kernel 3: projections ----------
// grid (N/64=16, 20 mtiles, B), block 32 (one wave). Wave: 16 rows x 64 cols.
__global__ __launch_bounds__(32) void k_proj(
    const U16* Wstack, const U16* ht, const U16* mt_,
    const float* bq, const float* bk, const float* bv,
    const float* bk2, const float* bv2,
    U16* Qt, U16* Kth, U16* Ktm, U16* Vh, U16* Vm) {
  int lane = threadIdx.x;
  int n0 = blockIdx.x * 64;
  int mtile = blockIdx.y;        // 0..19
  int b = blockIdx.z;
  int grp = mtile >> 2;          // 0:Wq 1:Wk 2:Wv 3:Wk2 4:Wv2
  const U16* X = ((grp < 3) ? ht : mt_) + (size_t)b * (1024 * 64);
  const U16* Arow = Wstack + (mtile * 16) * 64;

  v8f acc[4] = {zero8(), zero8(), zero8(), zero8()};
#pragma unroll
  for (int ks = 0; ks < 2; ++ks) {
    v16bf a = load_a_frag(Arow + ks * 32, 64, lane);
#pragma unroll
    for (int j = 0; j < 4; ++j) {
      v16bf bb = load_b_frag(X + (size_t)(n0 + j * 16) * 64 + ks * 32, 64, lane);
      acc[j] = wmma_bf16(a, bb, acc[j]);
    }
  }

  const float* bias = (grp == 0) ? bq : (grp == 1) ? bk : (grp == 2) ? bv
                      : (grp == 3) ? bk2 : bv2;
  int rbase = (mtile - grp * 4) * 16;   // row base within its 64-row group
  int half = lane >> 4, col = lane & 15;
#pragma unroll
  for (int j = 0; j < 4; ++j) {
#pragma unroll
    for (int v = 0; v < 8; ++v) {
      int r = rbase + half * 8 + v;     // 0..63 within group
      int n = n0 + j * 16 + col;
      U16 val = f2bfu(acc[j][v] + bias[r]);
      size_t tr = (size_t)((b << 10) + n) * 64 + r;       // transposed (N,64)
      size_t nat = ((size_t)b * 64 + r) * 1024 + n;       // natural (64,N)
      switch (grp) {
        case 0: Qt[tr] = val; break;
        case 1: Kth[tr] = val; break;
        case 3: Ktm[tr] = val; break;
        case 2: Vh[nat] = val; break;
        default: Vm[nat] = val; break;
      }
    }
  }
}

// ---------- kernel 4: attention (per batch, branch, 16-query tile) ----------
// block 128 (4 waves). S(16x1024 f32, 64KB) + P(16x1024 bf16, 32KB) in LDS.
__global__ __launch_bounds__(128) void k_attn(
    const U16* Qt, const U16* Kth, const U16* Ktm,
    const U16* Vh, const U16* Vm, U16* Zt) {
  __shared__ float S[16 * 1024];
  __shared__ U16 P[16 * 1024];
  __shared__ float red[128];
  __shared__ float rowmax[16];
  __shared__ float rowsum[16];

  int tid = threadIdx.x;
  int lane = tid & 31, w = tid >> 5;
  int q0 = blockIdx.x * 16;
  int branch = blockIdx.y;
  int b = blockIdx.z;

  const U16* Kt = (branch ? Ktm : Kth) + (size_t)b * (1024 * 64);
  const U16* V  = (branch ? Vm : Vh) + (size_t)b * (64 * 1024);
  const U16* Qb = Qt + (size_t)((b << 10) + q0) * 64;

  int half = lane >> 4, col = lane & 15;

  // Phase 1: S = Q^T K ; wave w covers keys [w*256, w*256+256)
  v16bf a0 = load_a_frag(Qb, 64, lane);        // K-dim 0..31
  v16bf a1 = load_a_frag(Qb + 32, 64, lane);   // K-dim 32..63
  for (int t = 0; t < 16; ++t) {
    int key0 = w * 256 + t * 16;
    v8f acc = zero8();
    acc = wmma_bf16(a0, load_b_frag(Kt + (size_t)key0 * 64, 64, lane), acc);
    acc = wmma_bf16(a1, load_b_frag(Kt + (size_t)key0 * 64 + 32, 64, lane), acc);
#pragma unroll
    for (int v = 0; v < 8; ++v)
      S[((half * 8 + v) << 10) + key0 + col] = acc[v];
  }

  // Warm caches for the V panel (needed in phase 3, ~2 phases from now).
  // 128 threads x 8 lines x 128B = the whole 128KB panel.
#pragma unroll
  for (int pf = 0; pf < 8; ++pf)
    __builtin_prefetch((const char*)V + (size_t)tid * 1024 + pf * 128, 0, 0);

  __syncthreads();

  // Phase 2: softmax over 1024 keys per row. 8 threads per row.
  {
    int row = tid & 15, seg = tid >> 4;
    const float* Sr = S + (row << 10) + seg * 128;
    float mx = -3.4e38f;
    for (int j = 0; j < 128; ++j) mx = fmaxf(mx, Sr[j]);
    red[tid] = mx;
    __syncthreads();
    if (tid < 16) {
      float m2 = red[tid];
      for (int s = 1; s < 8; ++s) m2 = fmaxf(m2, red[tid + s * 16]);
      rowmax[tid] = m2;
    }
    __syncthreads();
    float rm = rowmax[row];
    float sm = 0.f;
    for (int j = 0; j < 128; ++j) sm += __expf(Sr[j] - rm);
    red[tid] = sm;
    __syncthreads();
    if (tid < 16) {
      float s2 = 0.f;
      for (int s = 0; s < 8; ++s) s2 += red[tid + s * 16];
      rowsum[tid] = s2;
    }
    __syncthreads();
  }
  for (int idx = tid; idx < 16 * 1024; idx += 128) {
    int row = idx >> 10;
    float p = __expf(S[idx] - rowmax[row]) / rowsum[row];
    P[idx] = f2bfu(p);
  }
  __syncthreads();

  // Phase 3: Z(16x64) = P(16x1024) @ V^T(1024x64); wave w -> channels [w*16, w*16+16)
  int c0 = w * 16;
  v8f acc = zero8();
#pragma unroll 4
  for (int ks = 0; ks < 32; ++ks) {
    v16bf pa = load_a_frag(P + ks * 32, 1024, lane);                  // LDS A-frag
    v16bf vb = load_b_frag(V + (size_t)c0 * 1024 + ks * 32, 1024, lane);
    acc = wmma_bf16(pa, vb, acc);
  }
#pragma unroll
  for (int v = 0; v < 8; ++v) {
    int n = q0 + half * 8 + v;
    Zt[(size_t)((b << 10) + n) * 128 + branch * 64 + c0 + col] = f2bfu(acc[v]);
  }
}

// ---------- kernel 5: Z = Wz @ [Z_h;Z_m] + bz, stored transposed into Xfin cols 0..127 ----------
__global__ __launch_bounds__(32) void k_zmix(
    const U16* Wz16, const float* bz, const U16* Zt, U16* Xfin) {
  int lane = threadIdx.x;
  int n0 = blockIdx.x * 64;
  int mtile = blockIdx.y;   // 0..7
  int b = blockIdx.z;
  const U16* Arow = Wz16 + mtile * 16 * 128;
  const U16* X = Zt + ((size_t)b << 10) * 128;

  v8f acc[4] = {zero8(), zero8(), zero8(), zero8()};
#pragma unroll
  for (int ks = 0; ks < 4; ++ks) {
    v16bf a = load_a_frag(Arow + ks * 32, 128, lane);
#pragma unroll
    for (int j = 0; j < 4; ++j) {
      v16bf bb = load_b_frag(X + (size_t)(n0 + j * 16) * 128 + ks * 32, 128, lane);
      acc[j] = wmma_bf16(a, bb, acc[j]);
    }
  }
  int half = lane >> 4, col = lane & 15;
#pragma unroll
  for (int j = 0; j < 4; ++j) {
#pragma unroll
    for (int v = 0; v < 8; ++v) {
      int r = mtile * 16 + half * 8 + v;     // 0..127
      int n = n0 + j * 16 + col;
      Xfin[(size_t)((b << 10) + n) * 192 + r] = f2bfu(acc[j][v] + bz[r]);
    }
  }
}

// ---------- kernel 6: combined = Wm @ [Z;h] + bm, fused gating ----------
// grid (N/16=64, B), block 384 = 12 waves; each wave one 16-row strip of the
// 192 rows. The shared 16x192 bf16 activation panel is staged into LDS once
// via the Tensor Data Mover (TDM) instead of 12 redundant global reads.
__global__ __launch_bounds__(384) void k_final(
    const U16* Wm16, const float* bm, const U16* Xfin, const float* m_in,
    float* out_h, float* out_m) {
  __shared__ float comb[192 * 16];
  __shared__ U16 Xs[16 * 192];     // 6KB staged activation panel
  int tid = threadIdx.x, lane = tid & 31, w = tid >> 5;  // w: 0..11
  int n0 = blockIdx.x * 16;
  int b = blockIdx.y;
  const U16* Arow = Wm16 + w * 16 * 192;
  const U16* X = Xfin + (size_t)((b << 10) + n0) * 192;

#if __has_builtin(__builtin_amdgcn_tensor_load_to_lds) && __has_builtin(__builtin_amdgcn_s_wait_tensorcnt)
  if (w == 0) {
    // Tensor DMA descriptor (D#), 1-D tile of 3072 bf16 elements.
    unsigned lds_addr = (unsigned)(size_t)(void*)Xs;
    unsigned long long ga = (unsigned long long)(size_t)X;  // global byte addr
    const unsigned td0 = 16 * 192;     // tensor_dim0 (elements)
    const unsigned tile0 = 16 * 192;   // tile_dim0 (elements)
    uint4v g0;
    g0[0] = 1u;                                          // count=1 (valid)
    g0[1] = lds_addr;                                    // lds_addr
    g0[2] = (unsigned)(ga & 0xFFFFFFFFu);                // global_addr[31:0]
    g0[3] = (unsigned)((ga >> 32) & 0x1FFFFFFu) | (2u << 30);  // [56:32] | type=2
    int8v g1;
    g1[0] = (int)(1u << 16);                 // data_size=1 (2 bytes)
    g1[1] = (int)((td0 & 0xFFFFu) << 16);    // tensor_dim0[15:0] at bits 63:48
    g1[2] = (int)(td0 >> 16);                // tensor_dim0[31:16]; tensor_dim1=0
    g1[3] = (int)(tile0 << 16);              // tile_dim0 at bits 127:112
    g1[4] = 0;                               // tile_dim1=0 (unused), tile_dim2=0
    g1[5] = (int)td0;                        // tensor_dim0_stride[31:0]
    g1[6] = 0;
    g1[7] = 0;
    int4v g2 = {0, 0, 0, 0};
    int4v g3 = {0, 0, 0, 0};
    int8v g4 = {0, 0, 0, 0, 0, 0, 0, 0};     // 6-arg toolchain: extra group
    __builtin_amdgcn_tensor_load_to_lds(g0, g1, g2, g3, g4, 0);
    __builtin_amdgcn_s_wait_tensorcnt(0);
  }
#else
  for (int idx = tid; idx < 16 * 192; idx += 384) Xs[idx] = X[idx];
#endif
  __syncthreads();

  v8f acc = zero8();
#pragma unroll
  for (int ks = 0; ks < 6; ++ks) {
    v16bf a = load_a_frag(Arow + ks * 32, 192, lane);
    v16bf bb = load_b_frag(Xs + ks * 32, 192, lane);   // B-frags from LDS
    acc = wmma_bf16(a, bb, acc);
  }
  int half = lane >> 4, col = lane & 15;
#pragma unroll
  for (int v = 0; v < 8; ++v) {
    int r = w * 16 + half * 8 + v;
    comb[r * 16 + col] = acc[v] + bm[r];
  }
  __syncthreads();

  for (int idx = tid; idx < 1024; idx += 384) {
    int c = idx >> 4, j = idx & 15;
    int n = n0 + j;
    float mo = comb[c * 16 + j];
    float mg = comb[(64 + c) * 16 + j];
    float mi = comb[(128 + c) * 16 + j];
    float sig_i = 1.f / (1.f + __expf(-mi));
    float mv = m_in[(size_t)b * 65536 + c * 1024 + n];
    float nm = (1.f - sig_i) * mv + sig_i * tanhf(mg);
    float nh = nm / (1.f + __expf(-mo));
    out_h[(size_t)b * 65536 + c * 1024 + n] = nh;
    out_m[(size_t)b * 65536 + c * 1024 + n] = nm;
  }
}

// ---------- launch ----------
extern "C" void kernel_launch(void* const* d_in, const int* in_sizes, int n_in,
                              void* d_out, int out_size, void* d_ws, size_t ws_size,
                              hipStream_t stream) {
  const float* h   = (const float*)d_in[0];
  const float* m   = (const float*)d_in[1];
  const float* Wq  = (const float*)d_in[2];
  const float* bq  = (const float*)d_in[3];
  const float* Wk  = (const float*)d_in[4];
  const float* bk  = (const float*)d_in[5];
  const float* Wk2 = (const float*)d_in[6];
  const float* bk2 = (const float*)d_in[7];
  const float* Wv  = (const float*)d_in[8];
  const float* bv  = (const float*)d_in[9];
  const float* Wv2 = (const float*)d_in[10];
  const float* bv2 = (const float*)d_in[11];
  const float* Wz  = (const float*)d_in[12];
  const float* bz  = (const float*)d_in[13];
  const float* Wm  = (const float*)d_in[14];
  const float* bm  = (const float*)d_in[15];

  char* ws = (char*)d_ws;
  size_t off = 0;
  auto take = [&](size_t bytes) -> char* {
    char* p = ws + off;
    off += (bytes + 255) & ~(size_t)255;
    return p;
  };
  U16* Wstack = (U16*)take((size_t)320 * 64 * 2);
  U16* Wz16   = (U16*)take((size_t)128 * 128 * 2);
  U16* Wm16   = (U16*)take((size_t)192 * 192 * 2);
  U16* ht     = (U16*)take((size_t)16 * 1024 * 64 * 2);
  U16* mt_    = (U16*)take((size_t)16 * 1024 * 64 * 2);
  U16* Qt     = (U16*)take((size_t)16 * 1024 * 64 * 2);
  U16* Kth    = (U16*)take((size_t)16 * 1024 * 64 * 2);
  U16* Ktm    = (U16*)take((size_t)16 * 1024 * 64 * 2);
  U16* Vh     = (U16*)take((size_t)16 * 1024 * 64 * 2);
  U16* Vm     = (U16*)take((size_t)16 * 1024 * 64 * 2);
  U16* Zt     = (U16*)take((size_t)16 * 1024 * 128 * 2);
  U16* Xfin   = (U16*)take((size_t)16 * 1024 * 192 * 2);

  float* out_h = (float*)d_out;
  float* out_m = out_h + (size_t)16 * 64 * 1024;

  k_prep_weights<<<288, 256, 0, stream>>>(Wq, Wk, Wv, Wk2, Wv2, Wz, Wm,
                                          Wstack, Wz16, Wm16);
  k_transpose_in<<<8192, 256, 0, stream>>>(h, m, ht, mt_, Xfin);
  k_proj<<<dim3(16, 20, 16), 32, 0, stream>>>(Wstack, ht, mt_,
                                              bq, bk, bv, bk2, bv2,
                                              Qt, Kth, Ktm, Vh, Vm);
  k_attn<<<dim3(64, 2, 16), 128, 0, stream>>>(Qt, Kth, Ktm, Vh, Vm, Zt);
  k_zmix<<<dim3(16, 8, 16), 32, 0, stream>>>(Wz16, bz, Zt, Xfin);
  k_final<<<dim3(64, 16), 384, 0, stream>>>(Wm16, bm, Xfin, m, out_h, out_m);
}
